// LieNet_65678639891008
// MI455X (gfx1250) — matrix operand order repaired and checked by reading, not compiled, not run on hardware
//
#include <hip/hip_runtime.h>

#define S_DIM 2048
#define B_DIM 256
#define D_DIM 128
#define BT    64                 // batch rows per workgroup
#define NCOL  (B_DIM * D_DIM)    // 32768 independent scan columns
#define SM1   (S_DIM - 1)        // 2047 output steps
#define NCHUNK 16
#define CHLEN  128               // 16*128 >= 2047

typedef __attribute__((ext_vector_type(16))) __bf16 v16bf;
typedef __attribute__((ext_vector_type(8)))  float  v8f;
typedef __attribute__((ext_vector_type(4)))  int    v4i;

#define GLOBAL_AS __attribute__((address_space(1)))
#define LDS_AS    __attribute__((address_space(3)))

#if defined(__AMDGCN__) && __has_builtin(__builtin_amdgcn_global_load_async_to_lds_b128)
#define HAVE_ASYNC_LDS 1
#else
#define HAVE_ASYNC_LDS 0
#endif

__device__ __forceinline__ void wait_asynccnt0() {
#if defined(__AMDGCN__)
#if __has_builtin(__builtin_amdgcn_s_wait_asynccnt)
  __builtin_amdgcn_s_wait_asynccnt(0);
#else
  asm volatile("s_wait_asynccnt 0x0" ::: "memory");
#endif
#endif
}

// ---------------------------------------------------------------------------
// Kernel 0: repack Wm (128x128) and Wb (256x128) f32 -> bf16 in WMMA
// B-fragment order. For K-tile kt, N-tile nt, lane l, the 16 bf16 values are
// W[kt*32 + (l>>4)*16 + j][nt*16 + (l&15)], j=0..15  (32B contiguous per lane)
// ---------------------------------------------------------------------------
__global__ __launch_bounds__(256) void repack_weights(
    const float* __restrict__ Wm, const float* __restrict__ Wb,
    __bf16* __restrict__ wmP, __bf16* __restrict__ wbP)
{
  int idx = blockIdx.x * 256 + threadIdx.x;        // 0..49151
  if (idx < 16384) {                               // Wm: 4 kt * 8 nt * 32 lanes * 16
    int e = idx;
    int j = e & 15, lane = (e >> 4) & 31, nt = (e >> 9) & 7, kt = e >> 12;
    int n = nt * 16 + (lane & 15);
    int k = kt * 32 + ((lane >> 4) << 4) + j;
    wmP[e] = (__bf16)Wm[k * D_DIM + n];
  } else if (idx < 49152) {                        // Wb: 8 kt * 8 nt * 32 lanes * 16
    int e = idx - 16384;
    int j = e & 15, lane = (e >> 4) & 31, nt = (e >> 9) & 7, kt = e >> 12;
    int n = nt * 16 + (lane & 15);
    int k = kt * 32 + ((lane >> 4) << 4) + j;
    wbP[e] = (__bf16)Wb[k * D_DIM + n];
  }
}

// ---------------------------------------------------------------------------
// Kernel 1: fused  h[s],h[s+1] = relu(src@Wm+bm);
//                  u[s] = h[s] + relu([h[s],h[s+1]]@Wb + bb)   -> d_out
// One workgroup per (s, 64-row tile). 256 threads = 8 waves.
// Wave w: M-strip (w&3)*16, N-tiles (w>>2)*4 .. +3  (covers 64x128 tile).
// ---------------------------------------------------------------------------
__global__ __launch_bounds__(256) void fused_map_bracket(
    const float* __restrict__ src,
    const __bf16* __restrict__ wmP, const __bf16* __restrict__ wbP,
    const float* __restrict__ bm, const float* __restrict__ bb,
    float* __restrict__ out)
{
  __shared__ float  sA[2][BT][D_DIM];   // 64 KB : src[s], src[s+1] tiles (f32)
  __shared__ __bf16 sH[2][BT][D_DIM];   // 32 KB : h[s], h[s+1] tiles (bf16)

  const int s      = blockIdx.x;        // 0..S-2
  const int b0     = blockIdx.y * BT;
  const int tid    = threadIdx.x;
  const int lane   = tid & 31;
  const int wave   = tid >> 5;
  const int hi     = lane >> 4;
  const int lanelo = lane & 15;
  const int mstrip = (wave & 3) * 16;
  const int nbase  = (wave >> 2) * 4;

  // ---- stage both src slices to LDS (async DMA path on CDNA5) ----
  {
    const float4* g0 = reinterpret_cast<const float4*>(src + ((size_t)s * B_DIM + b0) * D_DIM);
    const float4* g1 = reinterpret_cast<const float4*>(src + ((size_t)(s + 1) * B_DIM + b0) * D_DIM);
    float4* l0 = reinterpret_cast<float4*>(&sA[0][0][0]);
    float4* l1 = reinterpret_cast<float4*>(&sA[1][0][0]);
#if HAVE_ASYNC_LDS
    #pragma unroll
    for (int i = 0; i < 8; ++i) {
      __builtin_amdgcn_global_load_async_to_lds_b128(
          (GLOBAL_AS v4i*)(g0 + tid + i * 256),
          (LDS_AS v4i*)(l0 + tid + i * 256), 0, 0);
      __builtin_amdgcn_global_load_async_to_lds_b128(
          (GLOBAL_AS v4i*)(g1 + tid + i * 256),
          (LDS_AS v4i*)(l1 + tid + i * 256), 0, 0);
    }
#else
    #pragma unroll
    for (int i = 0; i < 8; ++i) {
      l0[tid + i * 256] = g0[tid + i * 256];
      l1[tid + i * 256] = g1[tid + i * 256];
    }
#endif
    if (s + 2 < S_DIM)   // warm L2 for the workgroups handling later slices
      __builtin_prefetch(src + ((size_t)(s + 2) * B_DIM + b0) * D_DIM + tid * 32, 0, 1);
  }
#if HAVE_ASYNC_LDS
  wait_asynccnt0();
#endif
  __syncthreads();

  const int row = mstrip + lanelo;      // A-fragment row for this lane

  // ---- GEMM1 (x2 slices): h = relu(src @ Wm + bm) ----
  // B fragments of Wm are slice-independent: load once per kc, feed both slices.
  v8f acc[2][4] = {};
  #pragma unroll
  for (int kc = 0; kc < 4; ++kc) {
    v16bf bfr[4];
    #pragma unroll
    for (int t = 0; t < 4; ++t)
      bfr[t] = *reinterpret_cast<const v16bf*>(
          wmP + (((kc * 8) + (nbase + t)) * 32 + lane) * 16);
    #pragma unroll
    for (int ss = 0; ss < 2; ++ss) {
      // A fragment: contiguous k order, same as the wmP packing (lanes 0-15: K 0..15; 16-31: K 16..31)
      const float* ap = &sA[ss][row][kc * 32 + hi * 16];
      v16bf a;
      #pragma unroll
      for (int j = 0; j < 8; ++j) {
        a[j]     = (__bf16)ap[j];
        a[8 + j] = (__bf16)ap[8 + j];
      }
      #pragma unroll
      for (int t = 0; t < 4; ++t)
        acc[ss][t] = __builtin_amdgcn_wmma_f32_16x16x32_bf16(
            false, a, false, bfr[t], (short)0, acc[ss][t], false, false);
    }
  }

  // bias + relu; stash bf16 h for GEMM2; keep f32 h[s] for the final add
  v8f hs0[4];
  #pragma unroll
  for (int ss = 0; ss < 2; ++ss) {
    #pragma unroll
    for (int t = 0; t < 4; ++t) {
      const int col = (nbase + t) * 16 + lanelo;
      const float bv = bm[col];
      #pragma unroll
      for (int v = 0; v < 8; ++v) {
        float hv = acc[ss][t][v] + bv;
        hv = hv > 0.f ? hv : 0.f;
        acc[ss][t][v] = hv;
        sH[ss][mstrip + v + hi * 8][col] = (__bf16)hv;   // C-layout: M=v(+8), N=lane
      }
      if (ss == 0) hs0[t] = acc[ss][t];
    }
  }
  __syncthreads();

  // ---- GEMM2: tmp = relu([h[s],h[s+1]] @ Wb + bb), K = 256 ----
  v8f acc2[4] = {};
  #pragma unroll
  for (int kk = 0; kk < 8; ++kk) {
    const int ss = kk >> 2;
    const int kc = kk & 3;
    const __bf16* hp = &sH[ss][row][kc * 32 + hi * 16];
    v16bf a;
    #pragma unroll
    for (int j = 0; j < 8; ++j) { a[j] = hp[j]; a[8 + j] = hp[8 + j]; }
    #pragma unroll
    for (int t = 0; t < 4; ++t) {
      const v16bf b = *reinterpret_cast<const v16bf*>(
          wbP + (((kk * 8) + (nbase + t)) * 32 + lane) * 16);
      acc2[t] = __builtin_amdgcn_wmma_f32_16x16x32_bf16(
          false, a, false, b, (short)0, acc2[t], false, false);
    }
  }

  // ---- u = h[s] + relu(tmp + bb) -> d_out (pre-scan values) ----
  #pragma unroll
  for (int t = 0; t < 4; ++t) {
    const int col = (nbase + t) * 16 + lanelo;
    const float bv = bb[col];
    #pragma unroll
    for (int v = 0; v < 8; ++v) {
      float tv = acc2[t][v] + bv;
      tv = tv > 0.f ? tv : 0.f;
      const int r = mstrip + v + hi * 8;
      out[((size_t)s * B_DIM + (b0 + r)) * D_DIM + col] = hs0[t][v] + tv;
    }
  }
}

// ---------------------------------------------------------------------------
// Kernels 2-4: in-place cumsum along S over d_out (32768 independent columns)
// ---------------------------------------------------------------------------
__global__ __launch_bounds__(256) void scan_chunksum(
    const float* __restrict__ out, float* __restrict__ cs)
{
  const int chunk = blockIdx.x;
  const int col = blockIdx.y * 256 + threadIdx.x;
  const int i0 = chunk * CHLEN;
  const int i1 = (i0 + CHLEN < SM1) ? (i0 + CHLEN) : SM1;
  float acc = 0.f;
  for (int i = i0; i < i1; ++i) acc += out[(size_t)i * NCOL + col];
  cs[(size_t)chunk * NCOL + col] = acc;
}

__global__ __launch_bounds__(256) void scan_exclusive(float* __restrict__ cs)
{
  const int col = blockIdx.x * 256 + threadIdx.x;
  float r = 0.f;
  #pragma unroll
  for (int c = 0; c < NCHUNK; ++c) {
    float t = cs[(size_t)c * NCOL + col];
    cs[(size_t)c * NCOL + col] = r;
    r += t;
  }
}

__global__ __launch_bounds__(256) void scan_add(
    float* __restrict__ out, const float* __restrict__ cs)
{
  const int chunk = blockIdx.x;
  const int col = blockIdx.y * 256 + threadIdx.x;
  float r = cs[(size_t)chunk * NCOL + col];
  const int i0 = chunk * CHLEN;
  const int i1 = (i0 + CHLEN < SM1) ? (i0 + CHLEN) : SM1;
  for (int i = i0; i < i1; ++i) {
    r += out[(size_t)i * NCOL + col];
    out[(size_t)i * NCOL + col] = r;
  }
}

// ---------------------------------------------------------------------------
extern "C" void kernel_launch(void* const* d_in, const int* in_sizes, int n_in,
                              void* d_out, int out_size, void* d_ws, size_t ws_size,
                              hipStream_t stream) {
  (void)in_sizes; (void)n_in; (void)out_size; (void)ws_size;
  const float* src = (const float*)d_in[0];   // [S,B,D]
  const float* Wm  = (const float*)d_in[1];   // [D,D]
  const float* bm  = (const float*)d_in[2];   // [D]
  const float* Wb  = (const float*)d_in[3];   // [2D,D]
  const float* bb  = (const float*)d_in[4];   // [D]
  float* out = (float*)d_out;                 // [S-1,B,D]

  char* ws = (char*)d_ws;
  __bf16* wmP = (__bf16*)ws;                  //  32 KB
  __bf16* wbP = (__bf16*)(ws + 32768);        //  64 KB
  float*  cs  = (float*)(ws + 98304);         //   2 MB chunk sums

  repack_weights<<<192, 256, 0, stream>>>(Wm, Wb, wmP, wbP);
  fused_map_bracket<<<dim3(SM1, B_DIM / BT), 256, 0, stream>>>(src, wmP, wbP, bm, bb, out);
  scan_chunksum<<<dim3(NCHUNK, NCOL / 256), 256, 0, stream>>>(out, cs);
  scan_exclusive<<<NCOL / 256, 256, 0, stream>>>(cs);
  scan_add<<<dim3(NCHUNK, NCOL / 256), 256, 0, stream>>>(out, cs);
}
